// LSTMModel_49022756716769
// MI455X (gfx1250) — matrix operand order repaired and checked by reading, not compiled, not run on hardware
//
#include <hip/hip_runtime.h>

typedef _Float16 f16;
typedef __attribute__((ext_vector_type(16))) _Float16 v16h;
typedef __attribute__((ext_vector_type(8)))  _Float16 v8h;
typedef __attribute__((ext_vector_type(8)))  float    v8f;

#define BATCH  64
#define MAXLEN 512
#define EMBED  512
#define HIDDEN 1024
#define NGATE  4096   // 4*HIDDEN
#define NTAG   10

// ---------------------------------------------------------------------------
// WMMA fragment loaders (CDNA5 ISA 7.12.2 layouts, wave32)
// ---------------------------------------------------------------------------
__device__ __forceinline__ v16h make_v16h(v8h lo, v8h hi) {
  union { v16h v; v8h h[2]; } u;
  u.h[0] = lo; u.h[1] = hi;
  return u.v;
}

// A fragment: 16(M) x 32(K) f16 from row-major source with row stride lda.
// Lane L<16 : row M=L,   VGPRs 0-3 = K 0..7,  VGPRs 4-7 = K 16..23
// Lane L>=16: row M=L-16, VGPRs 0-3 = K 8..15, VGPRs 4-7 = K 24..31
__device__ __forceinline__ v16h load_a_frag(const f16* __restrict__ A, size_t lda, int k0) {
  const int lane = threadIdx.x & 31;
  const int row  = lane & 15;
  const int half = lane >> 4;
  const f16* p = A + (size_t)row * lda + (size_t)(k0 + half * 8);
  v8h lo = *(const v8h*)p;          // K = k0 + half*8 .. +7
  v8h hi = *(const v8h*)(p + 16);   // K = k0 + 16 + half*8 .. +7
  return make_v16h(lo, hi);
}

// B fragment: 32(K) x 16(N) f16 from K-major weight matrix Wt[k][n], ld=NGATE.
// Lane L holds column N = n0 + L%16; elements 0-7 = K k0+8h..+7, 8-15 = K k0+16+8h..+7 (h = L/16).
__device__ __forceinline__ v16h load_b_frag(const f16* __restrict__ Wt, int k0, int n0) {
  const int lane = threadIdx.x & 31;
  const f16* p = Wt + (size_t)(k0 + (lane >> 4) * 8) * NGATE + n0 + (lane & 15);
  v16h b;
  for (int s = 0; s < 8; ++s) {
    b[s]     = p[(size_t)s * NGATE];
    b[s + 8] = p[(size_t)(s + 16) * NGATE];
  }
  return b;
}

__device__ __forceinline__ float sigmoidf_fast(float x) {
  return 1.0f / (1.0f + __expf(-x));
}

// ---------------------------------------------------------------------------
// Diagonal-pipelined LSTM step kernel.
// Launch s in [0, MAXLEN]: blocks 0..63 run layer0 @ t=s, blocks 64..127 run
// layer1 @ t=s-1 (independent work -> both layers in flight each launch).
// Per block: 16 hidden units, 4 waves (one gate each), 64 batch rows.
// gates(64 x 4096) = [x_t ; h_{t-1}] @ Wt  (+bias in LDS epilogue).
// K-loop: 2x-unrolled ping-pong double buffering (no register rotation movs,
// no WMMA->VALU WAR hazard on the just-read fragment registers).
// ---------------------------------------------------------------------------
__global__ __launch_bounds__(128)
void lstm_pipe_kernel(const f16* __restrict__ embx,
                      f16* __restrict__ h0seq, f16* __restrict__ h1seq,
                      const f16* __restrict__ Wt0, const f16* __restrict__ Wt1,
                      const float* __restrict__ b0, const float* __restrict__ b1,
                      float* __restrict__ c0, float* __restrict__ c1,
                      int s)
{
  __shared__ float lds_g[4][64][16];

  const int layer = blockIdx.x >> 6;
  const int blk   = blockIdx.x & 63;
  const int t     = s - layer;
  if (t < 0 || t >= MAXLEN) return;   // uniform per block

  const f16*   Xt;   size_t ldx; int Kin;
  const f16*   Wt;   const float* bias; float* c_state; f16* Hseq;
  const size_t ldh = (size_t)MAXLEN * HIDDEN;
  if (layer == 0) {
    Xt  = embx + (size_t)t * EMBED;  ldx = (size_t)MAXLEN * EMBED;  Kin = EMBED;
    Wt  = Wt0; bias = b0; c_state = c0; Hseq = h0seq;
  } else {
    Xt  = h0seq + (size_t)t * HIDDEN; ldx = ldh;                    Kin = HIDDEN;
    Wt  = Wt1; bias = b1; c_state = c1; Hseq = h1seq;
  }
  const f16* Hprev = Hseq + (size_t)(t - 1) * HIDDEN;  // only read when t>0
  f16*       Hout  = Hseq + (size_t)t * HIDDEN;

  const int wave = threadIdx.x >> 5;
  const int lane = threadIdx.x & 31;
  const int hid0 = blk * 16;
  const int n0   = wave * HIDDEN + hid0;   // column in 4096-wide gate space

  // Ktot in {512, 1024, 1536, 2048} -> always a multiple of 64.
  const int Ktot = Kin + ((t > 0) ? HIDDEN : 0);

  // virtual concatenated A: rows from Xt for k<Kin, from Hprev for k>=Kin
  auto a_src = [&](int k0, const f16*& src, size_t& ld, int& kk) {
    if (k0 < Kin) { src = Xt;    ld = ldx; kk = k0; }
    else          { src = Hprev; ld = ldh; kk = k0 - Kin; }
  };

  v8f zero = {0.f,0.f,0.f,0.f,0.f,0.f,0.f,0.f};
  v8f acc0 = zero, acc1 = zero, acc2 = zero, acc3 = zero;

  // ping-pong fragment buffers
  v16h bA, aA0, aA1, aA2, aA3;
  v16h bB, aB0, aB1, aB2, aB3;

  // preload chunk 0 into buffer A
  {
    const f16* src; size_t ld; int kk;
    a_src(0, src, ld, kk);
    bA  = load_b_frag(Wt, 0, n0);
    aA0 = load_a_frag(src,                   ld, kk);
    aA1 = load_a_frag(src + (size_t)16 * ld, ld, kk);
    aA2 = load_a_frag(src + (size_t)32 * ld, ld, kk);
    aA3 = load_a_frag(src + (size_t)48 * ld, ld, kk);
  }

  for (int k0 = 0; k0 < Ktot; k0 += 64) {
    // ---- fill buffer B with chunk k0+32 (always exists: Ktot % 64 == 0) ----
    {
      const f16* src; size_t ld; int kk;
      a_src(k0 + 32, src, ld, kk);
      bB  = load_b_frag(Wt, k0 + 32, n0);
      aB0 = load_a_frag(src,                   ld, kk);
      aB1 = load_a_frag(src + (size_t)16 * ld, ld, kk);
      aB2 = load_a_frag(src + (size_t)32 * ld, ld, kk);
      aB3 = load_a_frag(src + (size_t)48 * ld, ld, kk);
    }
    if (k0 + 64 < Ktot)  // prefetch next chunk's weight rows into near caches
      __builtin_prefetch((const void*)(Wt + (size_t)(k0 + 64 + lane) * NGATE + n0), 0, 3);

    // ---- consume buffer A (loads above are to other registers) ----
    acc0 = __builtin_amdgcn_wmma_f32_16x16x32_f16(false, aA0, false, bA, (short)0, acc0, false, false);
    acc1 = __builtin_amdgcn_wmma_f32_16x16x32_f16(false, aA1, false, bA, (short)0, acc1, false, false);
    acc2 = __builtin_amdgcn_wmma_f32_16x16x32_f16(false, aA2, false, bA, (short)0, acc2, false, false);
    acc3 = __builtin_amdgcn_wmma_f32_16x16x32_f16(false, aA3, false, bA, (short)0, acc3, false, false);

    // ---- refill buffer A with chunk k0+64 while buffer B is consumed ----
    if (k0 + 64 < Ktot) {
      const f16* src; size_t ld; int kk;
      a_src(k0 + 64, src, ld, kk);
      bA  = load_b_frag(Wt, k0 + 64, n0);
      aA0 = load_a_frag(src,                   ld, kk);
      aA1 = load_a_frag(src + (size_t)16 * ld, ld, kk);
      aA2 = load_a_frag(src + (size_t)32 * ld, ld, kk);
      aA3 = load_a_frag(src + (size_t)48 * ld, ld, kk);
    }

    // ---- consume buffer B ----
    acc0 = __builtin_amdgcn_wmma_f32_16x16x32_f16(false, aB0, false, bB, (short)0, acc0, false, false);
    acc1 = __builtin_amdgcn_wmma_f32_16x16x32_f16(false, aB1, false, bB, (short)0, acc1, false, false);
    acc2 = __builtin_amdgcn_wmma_f32_16x16x32_f16(false, aB2, false, bB, (short)0, acc2, false, false);
    acc3 = __builtin_amdgcn_wmma_f32_16x16x32_f16(false, aB3, false, bB, (short)0, acc3, false, false);
  }

  // ---- spill this wave's gate block to LDS (C/D layout: lane=N, vgpr=M) ----
  {
    const int ncol = lane & 15;
    const int half = lane >> 4;
    v8f accs[4] = {acc0, acc1, acc2, acc3};
    #pragma unroll
    for (int mt = 0; mt < 4; ++mt) {
      #pragma unroll
      for (int v = 0; v < 8; ++v) {
        int m = mt * 16 + v + half * 8;
        lds_g[wave][m][ncol] = accs[mt][v];
      }
    }
  }
  __syncthreads();

  // ---- fused LSTM cell epilogue ----
  for (int idx = threadIdx.x; idx < 64 * 16; idx += 128) {
    const int m   = idx >> 4;        // batch row
    const int n   = idx & 15;
    const int hid = hid0 + n;
    float gi = lds_g[0][m][n] + bias[hid];
    float gf = lds_g[1][m][n] + bias[HIDDEN + hid];
    float gg = lds_g[2][m][n] + bias[2 * HIDDEN + hid];
    float go = lds_g[3][m][n] + bias[3 * HIDDEN + hid];

    float si = sigmoidf_fast(gi);
    float sf = sigmoidf_fast(gf);
    float so = sigmoidf_fast(go);
    float tg = tanhf(gg);

    const size_t ci = (size_t)m * HIDDEN + hid;
    float c = sf * c_state[ci] + si * tg;
    c_state[ci] = c;
    float h = so * tanhf(c);
    Hout[(size_t)m * ldh + hid] = (f16)h;
  }
}

// ---------------------------------------------------------------------------
// Preprocessing kernels
// ---------------------------------------------------------------------------
__global__ void embed_gather_kernel(const int* __restrict__ x,
                                    const float* __restrict__ emb,
                                    f16* __restrict__ embx)
{
  size_t idx = (size_t)blockIdx.x * blockDim.x + threadIdx.x;
  const size_t total = (size_t)BATCH * MAXLEN * EMBED;
  if (idx >= total) return;
  const int e = (int)(idx % EMBED);
  const size_t bt = idx / EMBED;
  const int tok = x[bt];
  embx[idx] = (f16)emb[(size_t)tok * EMBED + e];
}

// Build fused K-major f16 weights: Wt[k][n], k in [0, Kin+HIDDEN), n in [0, 4096)
// k < Kin  -> W_ih[n][k];  else -> W_hh[n][k-Kin]
__global__ void prep_weights_kernel(const float* __restrict__ Wih, int Kin,
                                    const float* __restrict__ Whh,
                                    f16* __restrict__ Wt)
{
  size_t idx = (size_t)blockIdx.x * blockDim.x + threadIdx.x;
  const size_t total = (size_t)(Kin + HIDDEN) * NGATE;
  if (idx >= total) return;
  const int n = (int)(idx % NGATE);
  const int k = (int)(idx / NGATE);
  float v = (k < Kin) ? Wih[(size_t)n * Kin + k]
                      : Whh[(size_t)n * HIDDEN + (k - Kin)];
  Wt[idx] = (f16)v;
}

__global__ void zero_f32_kernel(float* __restrict__ p, size_t n) {
  size_t i = (size_t)blockIdx.x * blockDim.x + threadIdx.x;
  if (i < n) p[i] = 0.0f;
}

__global__ void lens_kernel(const int* __restrict__ x, int* __restrict__ lens) {
  int b = threadIdx.x;
  if (b >= BATCH) return;
  int cnt = 0;
  for (int t = 0; t < MAXLEN; ++t) cnt += (x[(size_t)b * MAXLEN + t] == 0);
  lens[b] = MAXLEN - cnt;
}

// out[b][tag] = h1seq[b][lens[b]-1][:] . W_out[tag][:] + b_out[tag]
__global__ void output_kernel(const f16* __restrict__ h1seq,
                              const int* __restrict__ lens,
                              const float* __restrict__ W_out,
                              const float* __restrict__ b_out,
                              float* __restrict__ out)
{
  int idx = blockIdx.x * blockDim.x + threadIdx.x;
  if (idx >= BATCH * NTAG) return;
  const int b = idx / NTAG;
  const int tag = idx % NTAG;
  int tl = lens[b] - 1;
  if (tl < 0) tl = 0;
  const f16* h = h1seq + ((size_t)b * MAXLEN + tl) * HIDDEN;
  const float* w = W_out + (size_t)tag * HIDDEN;
  float s = b_out[tag];
  for (int k = 0; k < HIDDEN; ++k) s += (float)h[k] * w[k];
  out[(size_t)b * NTAG + tag] = s;
}

// ---------------------------------------------------------------------------
// Host launcher
// ---------------------------------------------------------------------------
extern "C" void kernel_launch(void* const* d_in, const int* in_sizes, int n_in,
                              void* d_out, int out_size, void* d_ws, size_t ws_size,
                              hipStream_t stream) {
  (void)in_sizes; (void)n_in; (void)out_size; (void)ws_size;
  const int*   x     = (const int*)  d_in[0];
  const float* emb   = (const float*)d_in[1];
  const float* W_ih0 = (const float*)d_in[2];
  const float* W_hh0 = (const float*)d_in[3];
  const float* b0    = (const float*)d_in[4];
  const float* W_ih1 = (const float*)d_in[5];
  const float* W_hh1 = (const float*)d_in[6];
  const float* b1    = (const float*)d_in[7];
  const float* W_out = (const float*)d_in[8];
  const float* b_out = (const float*)d_in[9];
  float* out = (float*)d_out;

  // --- workspace layout (all 256B aligned) ---
  char* ws = (char*)d_ws;
  size_t off = 0;
  auto alloc = [&](size_t bytes) -> void* {
    off = (off + 255) & ~(size_t)255;
    void* p = ws + off;
    off += bytes;
    return p;
  };
  f16*   Wt0   = (f16*)  alloc((size_t)(EMBED  + HIDDEN) * NGATE * sizeof(f16)); // 12 MB
  f16*   Wt1   = (f16*)  alloc((size_t)(HIDDEN + HIDDEN) * NGATE * sizeof(f16)); // 16 MB
  f16*   embx  = (f16*)  alloc((size_t)BATCH * MAXLEN * EMBED  * sizeof(f16));   // 32 MB
  f16*   h0seq = (f16*)  alloc((size_t)BATCH * MAXLEN * HIDDEN * sizeof(f16));   // 64 MB
  f16*   h1seq = (f16*)  alloc((size_t)BATCH * MAXLEN * HIDDEN * sizeof(f16));   // 64 MB
  float* c0    = (float*)alloc((size_t)BATCH * HIDDEN * sizeof(float));
  float* c1    = (float*)alloc((size_t)BATCH * HIDDEN * sizeof(float));
  int*   lens  = (int*)  alloc((size_t)BATCH * sizeof(int));

  // --- preprocessing (all parallel, on stream) ---
  {
    const size_t total = (size_t)BATCH * MAXLEN * EMBED;
    embed_gather_kernel<<<(unsigned)((total + 255) / 256), 256, 0, stream>>>(x, emb, embx);
  }
  {
    const size_t total = (size_t)(EMBED + HIDDEN) * NGATE;
    prep_weights_kernel<<<(unsigned)((total + 255) / 256), 256, 0, stream>>>(W_ih0, EMBED, W_hh0, Wt0);
  }
  {
    const size_t total = (size_t)(HIDDEN + HIDDEN) * NGATE;
    prep_weights_kernel<<<(unsigned)((total + 255) / 256), 256, 0, stream>>>(W_ih1, HIDDEN, W_hh1, Wt1);
  }
  {
    const size_t n = (size_t)BATCH * HIDDEN;
    zero_f32_kernel<<<(unsigned)((n + 255) / 256), 256, 0, stream>>>(c0, n);
    zero_f32_kernel<<<(unsigned)((n + 255) / 256), 256, 0, stream>>>(c1, n);
  }
  lens_kernel<<<1, 64, 0, stream>>>(x, lens);

  // --- diagonal-pipelined recurrence: 513 launches cover both layers ---
  for (int s = 0; s <= MAXLEN; ++s) {
    lstm_pipe_kernel<<<128, 128, 0, stream>>>(
        embx, h0seq, h1seq, Wt0, Wt1, b0, b1, c0, c1, s);
  }

  // --- final tap + output projection (64 x 10, K=1024) ---
  output_kernel<<<(BATCH * NTAG + 127) / 128, 128, 0, stream>>>(h1seq, lens, W_out, b_out, out);
}
